// LSTM_68788196213127
// MI455X (gfx1250) — compile-verified, matrix operand order checked by reading, not run on hardware
//
#include <hip/hip_runtime.h>
#include <stdint.h>

typedef unsigned short u16;
typedef __attribute__((ext_vector_type(16))) __bf16 v16bf;
typedef __attribute__((ext_vector_type(8)))  float  v8f;

#define NSEQ 2048   // B*P
#define LSEQ 32
#define GDIM 1024   // 4 gates x 256 (padded from 250)
#define HPAD 256

__device__ __forceinline__ u16 f2bf(float f) {
    uint32_t u = __float_as_uint(f);
    u += 0x7FFFu + ((u >> 16) & 1u);          // round-to-nearest-even
    return (u16)(u >> 16);
}
__device__ __forceinline__ float sigmoidf(float x) { return 1.0f / (1.0f + expf(-x)); }

// ---------------------------------------------------------------------------
// Build layer-0 input: [L][N][544] bf16 = concat(word 512, pos 4, dep 5, dir 1, pad)
// ---------------------------------------------------------------------------
__global__ __launch_bounds__(256) void embed_kernel(
    const float* __restrict__ word, const float* __restrict__ post,
    const float* __restrict__ dept, const float* __restrict__ dirt,
    const int* __restrict__ pid, const int* __restrict__ did,
    const int* __restrict__ rid, u16* __restrict__ x0)
{
    size_t idx = (size_t)blockIdx.x * 256 + threadIdx.x;
    const size_t total = (size_t)LSEQ * NSEQ * 544;
    if (idx >= total) return;
    int e = (int)(idx % 544);
    size_t ln = idx / 544;
    int n = (int)(ln % NSEQ);
    int l = (int)(ln / NSEQ);
    int tok = n * LSEQ + l;                    // [b,p,l] flat
    float v;
    if (e < 512)      v = word[(size_t)tok * 512 + e];
    else if (e < 516) v = post[pid[tok] * 4 + (e - 512)];
    else if (e < 521) v = dept[did[tok] * 5 + (e - 516)];
    else if (e == 521) v = dirt[rid[tok]];
    else v = 0.0f;
    x0[idx] = f2bf(v);
}

// ---------------------------------------------------------------------------
// Repack weights: Wcat[1024][Dpad+256] bf16 = [Wih | pad | Whh | pad], rows in
// padded-gate layout (gate*256 + r, r<250 valid). Also padded f32 bias.
// ---------------------------------------------------------------------------
__global__ __launch_bounds__(256) void repack_kernel(
    const float* __restrict__ Wih, const float* __restrict__ Whh,
    const float* __restrict__ bias, u16* __restrict__ Wcat,
    float* __restrict__ bpad, int D, int Dpad)
{
    int Ktot = Dpad + HPAD;
    int idx = blockIdx.x * 256 + threadIdx.x;
    if (idx >= GDIM * Ktot) return;
    int row = idx / Ktot;
    int k   = idx - row * Ktot;
    int gate = row >> 8;
    int r    = row & 255;
    bool valid = (r < 250);
    int src = gate * 250 + r;
    float v = 0.0f;
    if (valid) {
        if (k < D)                          v = Wih[(size_t)src * D + k];
        else if (k >= Dpad && k < Dpad+250) v = Whh[(size_t)src * 250 + (k - Dpad)];
    }
    Wcat[(size_t)row * Ktot + k] = f2bf(v);
    if (k == 0) bpad[row] = valid ? bias[src] : 0.0f;
}

__global__ __launch_bounds__(256) void zero_u16(u16* __restrict__ p, int n)
{
    int i = blockIdx.x * 256 + threadIdx.x;
    if (i < n) p[i] = 0;
}

// ---------------------------------------------------------------------------
// gates = [x_t | h] @ Wcat^T + bias    (WMMA bf16, f32 accumulate)
// A1: x_t [2048][lda1] bf16 (K1 = lda1),  A2: h [2048][256] bf16
// W: Wcat [1024][Ktot] bf16 (row = output col, i.e. B^T row-major; the h-part
//    of K is contiguous after the x-part, so B offsets stride straight thru)
// out: gates [2048][1024] f32
// Block = 128 threads (4 waves); block tile 64(M)x128(N); wave tile 32x64
// (2x4 = 8 WMMAs per K-step). Addressing = uniform base pointer (SGPR) +
// 32-bit per-lane offset (VGPR) + uniform k0 -> global_load_b128 saddr form.
// ---------------------------------------------------------------------------
__global__ __launch_bounds__(128) void gates_gemm(
    const u16* __restrict__ A1, int lda1,
    const u16* __restrict__ A2,
    const u16* __restrict__ W, int Ktot,
    const float* __restrict__ bias,
    float* __restrict__ out)
{
    union FragAB { v16bf v; uint4 q[2]; };
    union FragC  { v8f  v; float f[8]; };

    const int lane = threadIdx.x & 31;
    const int wave = threadIdx.x >> 5;
    const int half = lane >> 4;
    const int l15  = lane & 15;
    const int m0 = blockIdx.y * 64  + (wave >> 1) * 32;
    const int n0 = blockIdx.x * 128 + (wave & 1) * 64;
    const int K1 = lda1;

    FragC c[2][4];
#pragma unroll
    for (int j = 0; j < 4; ++j) {
        float bv = bias[n0 + j * 16 + l15];
#pragma unroll
        for (int i = 0; i < 2; ++i)
#pragma unroll
            for (int v = 0; v < 8; ++v) c[i][j].f[v] = bv;
    }

    // per-lane element offsets (32-bit, stay in VGPRs; bases stay in SGPRs)
    uint32_t aoff[2], hoff[2], boff[4];
#pragma unroll
    for (int i = 0; i < 2; ++i) {
        aoff[i] = (uint32_t)(m0 + i * 16 + l15) * (uint32_t)lda1 + 8u * half;
        hoff[i] = (uint32_t)(m0 + i * 16 + l15) * (uint32_t)HPAD + 8u * half;
    }
#pragma unroll
    for (int j = 0; j < 4; ++j)
        boff[j] = (uint32_t)(n0 + j * 16 + l15) * (uint32_t)Ktot + 8u * half;

    // ---- phase 1: x_t contribution (K1 = 544 or 512) ----
    for (int k0 = 0; k0 < K1; k0 += 32) {
        FragAB a[2], b[4];
#pragma unroll
        for (int i = 0; i < 2; ++i) {
            const u16* p = A1 + aoff[i] + k0;
            a[i].q[0] = *(const uint4*)(p);        // K = +8*half .. +7
            a[i].q[1] = *(const uint4*)(p + 16);   // K = +16+8*half .. +7
        }
#pragma unroll
        for (int j = 0; j < 4; ++j) {
            const u16* p = W + boff[j] + k0;
            b[j].q[0] = *(const uint4*)(p);
            b[j].q[1] = *(const uint4*)(p + 16);
        }
#pragma unroll
        for (int i = 0; i < 2; ++i)
#pragma unroll
            for (int j = 0; j < 4; ++j)
                c[i][j].v = __builtin_amdgcn_wmma_f32_16x16x32_bf16(
                    false, a[i].v, false, b[j].v, (short)0, c[i][j].v, false, false);
    }

    // ---- phase 2: recurrent h contribution (K = 256, fully unrolled) ----
#pragma unroll
    for (int k0 = 0; k0 < HPAD; k0 += 32) {
        FragAB a[2], b[4];
#pragma unroll
        for (int i = 0; i < 2; ++i) {
            const u16* p = A2 + hoff[i] + k0;
            a[i].q[0] = *(const uint4*)(p);
            a[i].q[1] = *(const uint4*)(p + 16);
        }
#pragma unroll
        for (int j = 0; j < 4; ++j) {
            const u16* p = W + boff[j] + K1 + k0;  // continues past x-part of K
            b[j].q[0] = *(const uint4*)(p);
            b[j].q[1] = *(const uint4*)(p + 16);
        }
#pragma unroll
        for (int i = 0; i < 2; ++i)
#pragma unroll
            for (int j = 0; j < 4; ++j)
                c[i][j].v = __builtin_amdgcn_wmma_f32_16x16x32_bf16(
                    false, a[i].v, false, b[j].v, (short)0, c[i][j].v, false, false);
    }

    // ---- store ----
#pragma unroll
    for (int i = 0; i < 2; ++i)
#pragma unroll
        for (int j = 0; j < 4; ++j)
#pragma unroll
            for (int v = 0; v < 8; ++v) {
                int row = m0 + i * 16 + v + 8 * half;
                int col = n0 + j * 16 + l15;
                out[(size_t)row * GDIM + col] = c[i][j].f[v];
            }
}

// ---------------------------------------------------------------------------
// LSTM cell elementwise: c,h update; h -> bf16 state + bf16 next-layer input
// ---------------------------------------------------------------------------
__global__ __launch_bounds__(256) void lstm_elem(
    const float* __restrict__ gates, float* __restrict__ cbuf,
    u16* __restrict__ hbf, u16* __restrict__ xout,      // xout pre-offset (t, dir)
    float* __restrict__ hfin,                           // pre-offset (dir)
    int first, int writeFin)
{
    int idx = blockIdx.x * 256 + threadIdx.x;           // over 2048*256
    int n = idx >> 8;
    int r = idx & 255;
    size_t g = (size_t)n * GDIM;
    float gi = gates[g + r];
    float gf = gates[g + 256 + r];
    float gg = gates[g + 512 + r];
    float go = gates[g + 768 + r];
    float cp = first ? 0.0f : cbuf[(size_t)n * HPAD + r];
    float cn = sigmoidf(gf) * cp + sigmoidf(gi) * tanhf(gg);
    float h  = sigmoidf(go) * tanhf(cn);
    cbuf[(size_t)n * HPAD + r] = cn;
    hbf[(size_t)n * HPAD + r] = f2bf(h);
    if (r < 250) {
        xout[(size_t)n * 512 + r] = f2bf(h);
        if (writeFin) hfin[(size_t)n * 512 + r] = h;
    }
}

// ---------------------------------------------------------------------------
// Readout: count-weighted path mean, concat(emb_x, path, emb_y) @ W_out^T + b,
// log_softmax over 5. One block per batch row.
// ---------------------------------------------------------------------------
__global__ __launch_bounds__(256) void final_kernel(
    const float* __restrict__ hfin, const float* __restrict__ counts,
    const float* __restrict__ embx, const float* __restrict__ emby,
    const float* __restrict__ Wout, const float* __restrict__ bout,
    float* __restrict__ out)
{
    int b = blockIdx.x, tid = threadIdx.x;
    float cw[16], csum = 0.0f;
#pragma unroll
    for (int p = 0; p < 16; ++p) { cw[p] = counts[b * 16 + p]; csum += cw[p]; }
    float inv = 1.0f / csum;

    float acc[5] = {0, 0, 0, 0, 0};
    for (int j = tid; j < 1524; j += 256) {
        float v;
        if (j < 512) v = embx[(size_t)b * 512 + j];
        else if (j < 1012) {
            int jj = j - 512;
            float s = 0.0f;
#pragma unroll
            for (int p = 0; p < 16; ++p)
                s += hfin[(size_t)(b * 16 + p) * 512 + jj] * cw[p];
            v = s * inv;
        } else v = emby[(size_t)b * 512 + (j - 1012)];
#pragma unroll
        for (int r = 0; r < 5; ++r) acc[r] += v * Wout[r * 1524 + j];
    }

    __shared__ float sm[5][256];
#pragma unroll
    for (int r = 0; r < 5; ++r) sm[r][tid] = acc[r];
    __syncthreads();
    for (int s = 128; s > 0; s >>= 1) {
        if (tid < s)
#pragma unroll
            for (int r = 0; r < 5; ++r) sm[r][tid] += sm[r][tid + s];
        __syncthreads();
    }
    if (tid == 0) {
        float lg[5], mx = -1e30f;
#pragma unroll
        for (int r = 0; r < 5; ++r) { lg[r] = sm[r][0] + bout[r]; mx = fmaxf(mx, lg[r]); }
        float se = 0.0f;
#pragma unroll
        for (int r = 0; r < 5; ++r) se += expf(lg[r] - mx);
        float lse = logf(se) + mx;
#pragma unroll
        for (int r = 0; r < 5; ++r) out[b * 5 + r] = lg[r] - lse;
    }
}

// ---------------------------------------------------------------------------
extern "C" void kernel_launch(void* const* d_in, const int* in_sizes, int n_in,
                              void* d_out, int out_size, void* d_ws, size_t ws_size,
                              hipStream_t stream)
{
    const float* word   = (const float*)d_in[0];
    const float* embx   = (const float*)d_in[1];
    const float* emby   = (const float*)d_in[2];
    const float* counts = (const float*)d_in[3];
    const float* post   = (const float*)d_in[4];
    const float* dept   = (const float*)d_in[5];
    const float* dirt   = (const float*)d_in[6];
    const float* Wih0f  = (const float*)d_in[7];
    const float* Whh0f  = (const float*)d_in[8];
    const float* b0f    = (const float*)d_in[9];
    const float* Wih0b  = (const float*)d_in[10];
    const float* Whh0b  = (const float*)d_in[11];
    const float* b0b    = (const float*)d_in[12];
    const float* WihF   = (const float*)d_in[13];
    const float* WhhF   = (const float*)d_in[14];
    const float* bF     = (const float*)d_in[15];
    const float* WihB   = (const float*)d_in[16];
    const float* WhhB   = (const float*)d_in[17];
    const float* bB     = (const float*)d_in[18];
    const float* Wout   = (const float*)d_in[19];
    const float* bout   = (const float*)d_in[20];
    const int*   pid    = (const int*)d_in[21];
    const int*   did    = (const int*)d_in[22];
    const int*   rid    = (const int*)d_in[23];

    // ---- carve workspace ----
    char* w = (char*)d_ws;
    auto carve = [&](size_t bytes) { void* p = (void*)w; w += (bytes + 255) & ~(size_t)255; return p; };
    u16*   x0    = (u16*)  carve((size_t)LSEQ * NSEQ * 544 * 2);
    u16*   xA    = (u16*)  carve((size_t)LSEQ * NSEQ * 512 * 2);
    u16*   xB    = (u16*)  carve((size_t)LSEQ * NSEQ * 512 * 2);
    float* gates = (float*)carve((size_t)NSEQ * GDIM * 4);
    u16*   hbuf  = (u16*)  carve((size_t)NSEQ * HPAD * 2);
    float* cbuf  = (float*)carve((size_t)NSEQ * HPAD * 4);
    float* hfin  = (float*)carve((size_t)NSEQ * 512 * 4);

    struct Spec { const float *Wih, *Whh, *b; int D, Dpad; };
    Spec sp[6] = {
        { Wih0f,               Whh0f,              b0f,        522, 544 },
        { Wih0b,               Whh0b,              b0b,        522, 544 },
        { WihF,                WhhF,               bF,         500, 512 },
        { WihB,                WhhB,               bB,         500, 512 },
        { WihF + 1000 * 500,   WhhF + 1000 * 250,  bF + 1000,  500, 512 },
        { WihB + 1000 * 500,   WhhB + 1000 * 250,  bB + 1000,  500, 512 },
    };
    u16*   Wcat[6];
    float* bpad[6];
    for (int ld = 0; ld < 6; ++ld) {
        int Ktot = sp[ld].Dpad + HPAD;
        Wcat[ld] = (u16*)  carve((size_t)GDIM * Ktot * 2);
        bpad[ld] = (float*)carve((size_t)GDIM * 4);
    }

    // ---- weight repack (bf16, padded gate layout) ----
    for (int ld = 0; ld < 6; ++ld) {
        int Ktot = sp[ld].Dpad + HPAD;
        int total = GDIM * Ktot;
        repack_kernel<<<(total + 255) / 256, 256, 0, stream>>>(
            sp[ld].Wih, sp[ld].Whh, sp[ld].b, Wcat[ld], bpad[ld], sp[ld].D, sp[ld].Dpad);
    }

    // ---- layer-0 input assembly ----
    {
        size_t total = (size_t)LSEQ * NSEQ * 544;
        embed_kernel<<<(unsigned)((total + 255) / 256), 256, 0, stream>>>(
            word, post, dept, dirt, pid, did, rid, x0);
    }

    // ---- 3 bidirectional LSTM layers ----
    const u16* xin = x0;
    int ldin = 544;
    u16* xouts[3] = { xA, xB, xA };
    for (int layer = 0; layer < 3; ++layer) {
        u16* xout = xouts[layer];
        for (int dir = 0; dir < 2; ++dir) {
            int ld = layer * 2 + dir;
            int Ktot = ldin + HPAD;
            zero_u16<<<(NSEQ * HPAD) / 256, 256, 0, stream>>>(hbuf, NSEQ * HPAD);
            for (int s = 0; s < LSEQ; ++s) {
                int t = (dir == 0) ? s : (LSEQ - 1 - s);
                gates_gemm<<<dim3(GDIM / 128, NSEQ / 64), 128, 0, stream>>>(
                    xin + (size_t)t * NSEQ * ldin, ldin, hbuf, Wcat[ld], Ktot,
                    bpad[ld], gates);
                int writeFin = (layer == 2 && t == LSEQ - 1) ? 1 : 0;
                lstm_elem<<<(NSEQ * HPAD) / 256, 256, 0, stream>>>(
                    gates, cbuf, hbuf,
                    xout + (size_t)t * NSEQ * 512 + dir * 250,
                    hfin + dir * 250, (s == 0) ? 1 : 0, writeFin);
            }
        }
        xin = xout;
        ldin = 512;
    }

    // ---- readout ----
    final_kernel<<<128, 256, 0, stream>>>(hfin, counts, embx, emby, Wout, bout,
                                          (float*)d_out);
}
